// EEGMI_RMSU_Model_4226247819914
// MI455X (gfx1250) — matrix operand order repaired and checked by reading, not compiled, not run on hardware
//
#include <hip/hip_runtime.h>
#include <hip/hip_bf16.h>
#include <math.h>

typedef __attribute__((ext_vector_type(16))) __bf16 v16bf;
typedef __attribute__((ext_vector_type(8)))  __bf16 v8bf;
typedef __attribute__((ext_vector_type(8)))  float  v8f;
typedef __attribute__((ext_vector_type(2)))  float  v2f;
typedef unsigned int u32;

__device__ __forceinline__ __bf16 bfbits(unsigned short u) {
    return __builtin_bit_cast(__bf16, u);
}

#define CCH 22
#define NBD 5
#define TT  1000
#define CB  110     // CCH*NBD
#define DD  550
#define SS  64
#define BB  64

// ---------------------------------------------------------------------------
// 1. grouped conv (3x2, pad 1,1, groups=22) : x[64,22,1000,4] -> y[64,110,1000,5]
// ---------------------------------------------------------------------------
__global__ __launch_bounds__(256)
void conv_kernel(const float* __restrict__ x, const float* __restrict__ cw,
                 const float* __restrict__ cb, float* __restrict__ y)
{
    size_t idx = (size_t)blockIdx.x * 256 + threadIdx.x;
    if (idx >= (size_t)BB * CB * TT * 5) return;
    int w = idx % 5;
    int t = (idx / 5) % TT;
    int o = (idx / (5 * TT)) % CB;
    int b = idx / ((size_t)5 * TT * CB);
    int c = o / NBD;
    float s = cb[o];
    #pragma unroll
    for (int kh = 0; kh < 3; ++kh) {
        int th = t + kh - 1;
        if (th < 0 || th >= TT) continue;
        #pragma unroll
        for (int kw = 0; kw < 2; ++kw) {
            int tw = w + kw - 1;
            if (tw < 0 || tw >= 4) continue;
            s += x[(((size_t)b * CCH + c) * TT + th) * 4 + tw] * cw[o * 6 + kh * 2 + kw];
        }
    }
    y[idx] = s;
}

// ---------------------------------------------------------------------------
// 2. avg over (T,5) per (b,channel):  y -> avg_c[64,110]
// ---------------------------------------------------------------------------
__global__ __launch_bounds__(256)
void avgc_kernel(const float* __restrict__ y, float* __restrict__ avg_c)
{
    __shared__ float red[256];
    int row = blockIdx.x;                // b*110+o
    const float* p = y + (size_t)row * (TT * 5);
    float s = 0.f;
    for (int i = threadIdx.x; i < TT * 5; i += 256) s += p[i];
    red[threadIdx.x] = s; __syncthreads();
    for (int st = 128; st > 0; st >>= 1) {
        if (threadIdx.x < st) red[threadIdx.x] += red[threadIdx.x + st];
        __syncthreads();
    }
    if (threadIdx.x == 0) avg_c[row] = red[0] * (1.f / (TT * 5));
}

// ---------------------------------------------------------------------------
// 3. spatial attention MLP + softmax -> wc[64,110]
// ---------------------------------------------------------------------------
__global__ __launch_bounds__(128)
void samlp_kernel(const float* __restrict__ avg_c,
                  const float* __restrict__ w1, const float* __restrict__ b1,
                  const float* __restrict__ w2, const float* __restrict__ b2,
                  float* __restrict__ wc)
{
    __shared__ float av[CB], hid[CB / 2], out[CB], red[128];
    int b = blockIdx.x, tid = threadIdx.x;
    if (tid < CB) av[tid] = avg_c[b * CB + tid];
    __syncthreads();
    if (tid < CB / 2) {
        float s = b1[tid];
        for (int i = 0; i < CB; ++i) s += av[i] * w1[i * (CB / 2) + tid];
        hid[tid] = fmaxf(s, 0.f);
    }
    __syncthreads();
    if (tid < CB) {
        float s = b2[tid];
        for (int j = 0; j < CB / 2; ++j) s += hid[j] * w2[j * CB + tid];
        out[tid] = s;
    }
    __syncthreads();
    red[tid] = (tid < CB) ? out[tid] : -1e30f; __syncthreads();
    for (int st = 64; st > 0; st >>= 1) {
        if (tid < st) red[tid] = fmaxf(red[tid], red[tid + st]);
        __syncthreads();
    }
    float mx = red[0]; __syncthreads();
    float e = (tid < CB) ? __expf(out[tid] - mx) : 0.f;
    red[tid] = e; __syncthreads();
    for (int st = 64; st > 0; st >>= 1) {
        if (tid < st) red[tid] += red[tid + st];
        __syncthreads();
    }
    if (tid < CB) wc[b * CB + tid] = e / red[0];
}

// ---------------------------------------------------------------------------
// 4. avg_t[b,t] = mean over (o,w) of y*wc
// ---------------------------------------------------------------------------
__global__ __launch_bounds__(256)
void avgt_kernel(const float* __restrict__ y, const float* __restrict__ wc,
                 float* __restrict__ avg_t)
{
    __shared__ float red[256];
    int r = blockIdx.x;                 // b*1000+t
    int b = r / TT, t = r % TT;
    float s = 0.f;
    for (int j = threadIdx.x; j < DD; j += 256) {
        int o = j / NBD, w = j % NBD;
        s += y[(((size_t)b * CB + o) * TT + t) * 5 + w] * wc[b * CB + o];
    }
    red[threadIdx.x] = s; __syncthreads();
    for (int st = 128; st > 0; st >>= 1) {
        if (threadIdx.x < st) red[threadIdx.x] += red[threadIdx.x + st];
        __syncthreads();
    }
    if (threadIdx.x == 0) avg_t[r] = red[0] * (1.f / DD);
}

// ---------------------------------------------------------------------------
// 5. temporal attention MLP + softmax -> wt[64,1000]
// ---------------------------------------------------------------------------
__global__ __launch_bounds__(256)
void tamlp_kernel(const float* __restrict__ avg_t,
                  const float* __restrict__ w1, const float* __restrict__ b1,
                  const float* __restrict__ w2, const float* __restrict__ b2,
                  float* __restrict__ wt)
{
    __shared__ float av[TT], hid[TT / 2], out[TT], red[256];
    int b = blockIdx.x, tid = threadIdx.x;
    for (int i = tid; i < TT; i += 256) av[i] = avg_t[b * TT + i];
    __syncthreads();
    for (int j = tid; j < TT / 2; j += 256) {
        float s = b1[j];
        for (int i = 0; i < TT; ++i) s += av[i] * w1[i * (TT / 2) + j];
        hid[j] = fmaxf(s, 0.f);
    }
    __syncthreads();
    for (int k = tid; k < TT; k += 256) {
        float s = b2[k];
        for (int j = 0; j < TT / 2; ++j) s += hid[j] * w2[j * TT + k];
        out[k] = s;
    }
    __syncthreads();
    float mx = -1e30f;
    for (int k = tid; k < TT; k += 256) mx = fmaxf(mx, out[k]);
    red[tid] = mx; __syncthreads();
    for (int st = 128; st > 0; st >>= 1) {
        if (tid < st) red[tid] = fmaxf(red[tid], red[tid + st]);
        __syncthreads();
    }
    mx = red[0]; __syncthreads();
    float sm = 0.f;
    for (int k = tid; k < TT; k += 256) sm += __expf(out[k] - mx);
    red[tid] = sm; __syncthreads();
    for (int st = 128; st > 0; st >>= 1) {
        if (tid < st) red[tid] += red[tid + st];
        __syncthreads();
    }
    sm = red[0];
    for (int k = tid; k < TT; k += 256)
        wt[b * TT + k] = __expf(out[k] - mx) / sm;
}

// ---------------------------------------------------------------------------
// 6. LayerNorm + eval-BN, one block per (t,b) row. mode 0: build feat from
//    y*wc*wt; mode 1: read seq directly. Writes f32 xn and bf16 copy.
// ---------------------------------------------------------------------------
__global__ __launch_bounds__(256)
void ln_kernel(const float* __restrict__ src, const float* __restrict__ wc,
               const float* __restrict__ wt,
               const float* __restrict__ lng, const float* __restrict__ lnb,
               const float* __restrict__ bng, const float* __restrict__ bnb,
               float* __restrict__ xn, __bf16* __restrict__ xnh, int mode)
{
    __shared__ float rowS[DD];
    __shared__ float r1[256], r2[256];
    int r = blockIdx.x;                 // t*64+b
    int t = r >> 6, b = r & 63;
    int tid = threadIdx.x;
    float s1 = 0.f, s2 = 0.f;
    for (int d = tid; d < DD; d += 256) {
        float v;
        if (mode == 0) {
            int o = d / NBD, w = d - NBD * o;
            v = src[(((size_t)b * CB + o) * TT + t) * 5 + w]
                * wc[b * CB + o] * wt[b * TT + t];
        } else {
            v = src[(size_t)r * DD + d];
        }
        rowS[d] = v; s1 += v; s2 += v * v;
    }
    r1[tid] = s1; r2[tid] = s2; __syncthreads();
    for (int st = 128; st > 0; st >>= 1) {
        if (tid < st) { r1[tid] += r1[tid + st]; r2[tid] += r2[tid + st]; }
        __syncthreads();
    }
    float mu = r1[0] * (1.f / DD);
    float var = r2[0] * (1.f / DD) - mu * mu;
    float rs = rsqrtf(var + 1e-5f);
    for (int d = tid; d < DD; d += 256) {
        float v = (rowS[d] - mu) * rs * lng[d] + lnb[d];
        v = v * bng[d] + bnb[d];
        xn[(size_t)r * DD + d]  = v;
        xnh[(size_t)r * DD + d] = (__bf16)v;
    }
}

// ---------------------------------------------------------------------------
// 7a. pack Wg|Wq|Wv|Bw -> bf16 [550,256] + fused bias vector [256]
// ---------------------------------------------------------------------------
__global__ __launch_bounds__(256)
void pack_wcat_kernel(const float* __restrict__ Wg, const float* __restrict__ Wq,
                      const float* __restrict__ Wv, const float* __restrict__ Bw,
                      const float* __restrict__ bWg, const float* __restrict__ bUg,
                      const float* __restrict__ bg, const float* __restrict__ bq,
                      const float* __restrict__ bv,
                      __bf16* __restrict__ wcath, float* __restrict__ biascat)
{
    int idx = blockIdx.x * 256 + threadIdx.x;
    if (idx < DD * 256) {
        int d = idx >> 8, c = idx & 255;
        float v;
        if      (c < 64)  v = Wg[d * SS + c];
        else if (c < 128) v = Wq[d * SS + (c - 64)];
        else if (c < 192) v = Wv[d * SS + (c - 128)];
        else              v = Bw[d * SS + (c - 192)];
        wcath[idx] = (__bf16)v;
    }
    if (idx < 256) {
        int c = idx;
        float bv_;
        if      (c < 64)  bv_ = bWg[c] + bUg[c] + bg[c];
        else if (c < 128) bv_ = bq[c - 64];
        else if (c < 192) bv_ = bv[c - 128];
        else              bv_ = 0.f;                 // pure xn@Bw column
        biascat[c] = bv_;
    }
}

// 7b. generic f32 -> bf16 pack (Wo)
__global__ __launch_bounds__(256)
void pack_bf16_kernel(const float* __restrict__ src, __bf16* __restrict__ dst, int n)
{
    int i = blockIdx.x * 256 + threadIdx.x;
    if (i < n) dst[i] = (__bf16)src[i];
}

// ---------------------------------------------------------------------------
// 8. bf16 WMMA GEMM: C[M,N] = A[M,K] @ B[K,N] (+bias); mode 1: sigmoid(.)+res
//    64x64 C block-tile, 8 waves x 2 (16x16) tiles, K-step 32.
//    B staged TRANSPOSED in LDS so every WMMA fragment is ds_load_b128s.
// ---------------------------------------------------------------------------
__global__ __launch_bounds__(256)
void gemm_bf16_kernel(const __bf16* __restrict__ A, const __bf16* __restrict__ B,
                      const float* __restrict__ bias, const float* __restrict__ res,
                      float* __restrict__ C, int M, int N, int K, int mode)
{
    __shared__ __bf16 As [64][40];   // K-major, row stride 80B (16B aligned)
    __shared__ __bf16 BsT[64][40];   // N-major (transposed), row stride 80B
    int tid = threadIdx.x, lane = tid & 31, wv = tid >> 5;
    int nBlocks = (N + 63) >> 6;
    int bm = (blockIdx.x / nBlocks) << 6;
    int bn = (blockIdx.x % nBlocks) << 6;
    int t0 = 2 * wv;
    int m0 = (t0 >> 2) << 4;
    int n0a = (t0 & 3) << 4, n0b = ((t0 + 1) & 3) << 4;
    int l15 = lane & 15, hi = lane >> 4;

    // per-thread staging coordinates (8 contiguous elements each)
    int ar = tid >> 2,        ac = (tid & 3) << 3;   // A: row 0..63, col 0,8,16,24
    int br = tid >> 3,        bc = (tid & 7) << 3;   // B: row 0..31, col 0,8,...,56

    v8f acc0 = {0.f,0.f,0.f,0.f,0.f,0.f,0.f,0.f};
    v8f acc1 = {0.f,0.f,0.f,0.f,0.f,0.f,0.f,0.f};

    bool fullN = (bn + 64 <= N);

    for (int k0 = 0; k0 < K; k0 += 32) {
        bool fullK = (k0 + 32 <= K);
        // ---- stage A 64x32 ----
        if (fullK) {
            const u32* g = (const u32*)(A + (size_t)(bm + ar) * K + k0 + ac);
            u32* l = (u32*)&As[ar][ac];
            l[0] = g[0]; l[1] = g[1]; l[2] = g[2]; l[3] = g[3];
        } else {
            #pragma unroll
            for (int j = 0; j < 8; ++j) {
                As[ar][ac + j] = (k0 + ac + j < K)
                    ? A[(size_t)(bm + ar) * K + k0 + ac + j] : bfbits(0);
            }
        }
        // ---- stage B 32x64 (transposed into BsT[n][k]) ----
        if (fullK && fullN) {
            const u32* g = (const u32*)(B + (size_t)(k0 + br) * N + bn + bc);
            #pragma unroll
            for (int j = 0; j < 4; ++j) {
                u32 w = g[j];
                BsT[bc + 2 * j][br]     = bfbits((unsigned short)(w & 0xffffu));
                BsT[bc + 2 * j + 1][br] = bfbits((unsigned short)(w >> 16));
            }
        } else {
            #pragma unroll
            for (int j = 0; j < 8; ++j) {
                __bf16 v = bfbits(0);
                if (k0 + br < K && bn + bc + j < N)
                    v = B[(size_t)(k0 + br) * N + bn + bc + j];
                BsT[bc + j][br] = v;
            }
        }
        __syncthreads();

        // ---- fragments: all contiguous 16B LDS loads ----
        v8bf alo = *(const v8bf*)&As[m0 + l15][hi * 8];
        v8bf ahi = *(const v8bf*)&As[m0 + l15][16 + hi * 8];
        v16bf af = __builtin_shufflevector(alo, ahi,
                     0,1,2,3,4,5,6,7,8,9,10,11,12,13,14,15);
        v8bf b0l = *(const v8bf*)&BsT[n0a + l15][hi * 8];
        v8bf b0h = *(const v8bf*)&BsT[n0a + l15][16 + hi * 8];
        v16bf bf0 = __builtin_shufflevector(b0l, b0h,
                     0,1,2,3,4,5,6,7,8,9,10,11,12,13,14,15);
        v8bf b1l = *(const v8bf*)&BsT[n0b + l15][hi * 8];
        v8bf b1h = *(const v8bf*)&BsT[n0b + l15][16 + hi * 8];
        v16bf bf1 = __builtin_shufflevector(b1l, b1h,
                     0,1,2,3,4,5,6,7,8,9,10,11,12,13,14,15);

        acc0 = __builtin_amdgcn_wmma_f32_16x16x32_bf16(false, af, false, bf0,
                                                       (short)0, acc0, false, false);
        acc1 = __builtin_amdgcn_wmma_f32_16x16x32_bf16(false, af, false, bf1,
                                                       (short)0, acc1, false, false);
        __syncthreads();
    }

    #pragma unroll
    for (int r = 0; r < 8; ++r) {
        int row = bm + m0 + r + hi * 8;
        int c0 = bn + n0a + l15;
        if (c0 < N) {
            float v = acc0[r] + bias[c0];
            if (mode == 1) v = 1.f / (1.f + __expf(-v)) + res[(size_t)row * N + c0];
            C[(size_t)row * N + c0] = v;
        }
        int c1 = bn + n0b + l15;
        if (c1 < N) {
            float v = acc1[r] + bias[c1];
            if (mode == 1) v = 1.f / (1.f + __expf(-v)) + res[(size_t)row * N + c1];
            C[(size_t)row * N + c1] = v;
        }
    }
}

// ---------------------------------------------------------------------------
// 9. RMSU sequential scan (one persistent workgroup, 16 waves).
//    State: h[64,64] in LDS (padded rows), a/mB in registers. Per step:
//      U = h@Ug via V_WMMA_F32_16X16X4_F32, then elementwise update.
//    Ug held transposed in LDS so both operand loads are ds_load_b64.
// ---------------------------------------------------------------------------
__global__ __launch_bounds__(512)
void scan_kernel(const float* __restrict__ proj,   // [T*64,256] G|Q|V|W
                 const float* __restrict__ Ug,     // [64,64]
                 const float* __restrict__ Arow,   // [64]
                 const float* __restrict__ Bb,     // [64]
                 __bf16* __restrict__ H2h)         // [T*64,64]
{
    __shared__ float hS[64][66];     // padded: avoid 16-way bank conflicts
    __shared__ float ugT[64][66];    // transposed Ug: ugT[n][k]
    __shared__ float uS[64][64];
    __shared__ float tA[64], bB[64];
    int tid = threadIdx.x, lane = tid & 31, wv = tid >> 5;
    for (int i = tid; i < 4096; i += 512) {
        int k = i >> 6, n = i & 63;
        hS[k][n]  = 0.f;
        ugT[n][k] = Ug[i];
    }
    if (tid < 64) { tA[tid] = tanhf(Arow[tid]); bB[tid] = Bb[tid]; }
    float aR[8], mR[8];
    #pragma unroll
    for (int j = 0; j < 8; ++j) { aR[j] = 0.f; mR[j] = 0.f; }
    int m0 = (wv >> 2) << 4, n0 = (wv & 3) << 4;
    int l15 = lane & 15, hi = lane >> 4;
    int b = tid >> 3, s0 = (tid & 7) * 8;
    __syncthreads();

    for (int t = 0; t < TT; ++t) {
        if (t + 1 < TT)
            __builtin_prefetch(proj + (size_t)(t + 1) * (64 * 256) + tid * 32, 0, 1);

        // ---- phase A: U = h @ Ug (fp32 WMMA, K=64 in steps of 4) ----
        v8f acc = {0.f,0.f,0.f,0.f,0.f,0.f,0.f,0.f};
        #pragma unroll
        for (int k0 = 0; k0 < 64; k0 += 4) {
            v2f a2 = *(const v2f*)&hS[m0 + l15][k0 + 2 * hi];
            v2f b2 = *(const v2f*)&ugT[n0 + l15][k0 + 2 * hi];
            acc = __builtin_amdgcn_wmma_f32_16x16x4_f32(false, a2, false, b2,
                                                        (short)0, acc, false, false);
        }
        #pragma unroll
        for (int r = 0; r < 8; ++r) uS[m0 + r + hi * 8][n0 + l15] = acc[r];
        __syncthreads();

        // ---- phase B: elementwise state update (8 elems / thread) ----
        const float* row = proj + (size_t)(t * 64 + b) * 256;
        #pragma unroll
        for (int j = 0; j < 8; ++j) {
            int s = s0 + j;
            float G = row[s], Q = row[64 + s], V = row[128 + s], W = row[192 + s];
            float g  = 1.f / (1.f + __expf(-(G + uS[b][s])));
            float a2v = Q * V + __expf(-0.1f * g) * aR[j];
            float dmuB = (W - mR[j]) * 0.01f;
            float h2 = tA[s] * hS[b][s] + (W + bB[s]) * (1.f - g)
                       + 0.05f * dmuB + a2v;
            mR[j] += dmuB;
            aR[j] = a2v;
            hS[b][s] = h2;
            H2h[(size_t)(t * 64 + b) * 64 + s] = (__bf16)h2;
        }
        __syncthreads();
    }
}

// ---------------------------------------------------------------------------
// 10. classifier split-K partial: acc[64,128] += flat[64,Kc] @ w1[Kc,128]
//     fp32 WMMA 16x16x4 (bandwidth-bound; avoids re-encoding 281MB weights)
// ---------------------------------------------------------------------------
__global__ __launch_bounds__(256)
void clf_partial_kernel(const float* __restrict__ seq,   // out1, [T,64,550]
                        const float* __restrict__ W1,    // [550000,128]
                        float* __restrict__ acc)         // [64,128]
{
    int tid = threadIdx.x, lane = tid & 31, wv = tid >> 5;
    int l15 = lane & 15, hi = lane >> 4;
    int n0 = wv * 16;
    size_t kbase = (size_t)blockIdx.x * 2000;
    v8f a0 = {0.f,0.f,0.f,0.f,0.f,0.f,0.f,0.f};
    v8f a1 = a0, a2v = a0, a3 = a0;
    for (int kk = 0; kk < 2000; kk += 4) {
        size_t k  = kbase + kk + 2 * hi;
        size_t k1 = k + 1;
        int t0 = (int)(k / DD),  d0 = (int)(k - (size_t)t0 * DD);
        int t1 = (int)(k1 / DD), d1 = (int)(k1 - (size_t)t1 * DD);
        size_t base0 = (size_t)t0 * (64 * DD) + d0;
        size_t base1 = (size_t)t1 * (64 * DD) + d1;
        v2f bfr;
        bfr[0] = W1[k  * 128 + n0 + l15];
        bfr[1] = W1[k1 * 128 + n0 + l15];
        v2f af;
        af[0] = seq[base0 + (size_t)(l15) * DD];
        af[1] = seq[base1 + (size_t)(l15) * DD];
        a0 = __builtin_amdgcn_wmma_f32_16x16x4_f32(false, af, false, bfr, (short)0, a0, false, false);
        af[0] = seq[base0 + (size_t)(16 + l15) * DD];
        af[1] = seq[base1 + (size_t)(16 + l15) * DD];
        a1 = __builtin_amdgcn_wmma_f32_16x16x4_f32(false, af, false, bfr, (short)0, a1, false, false);
        af[0] = seq[base0 + (size_t)(32 + l15) * DD];
        af[1] = seq[base1 + (size_t)(32 + l15) * DD];
        a2v = __builtin_amdgcn_wmma_f32_16x16x4_f32(false, af, false, bfr, (short)0, a2v, false, false);
        af[0] = seq[base0 + (size_t)(48 + l15) * DD];
        af[1] = seq[base1 + (size_t)(48 + l15) * DD];
        a3 = __builtin_amdgcn_wmma_f32_16x16x4_f32(false, af, false, bfr, (short)0, a3, false, false);
    }
    #pragma unroll
    for (int r = 0; r < 8; ++r) {
        int rr = r + hi * 8, cc = n0 + l15;
        atomicAdd(&acc[(0  + rr) * 128 + cc], a0[r]);
        atomicAdd(&acc[(16 + rr) * 128 + cc], a1[r]);
        atomicAdd(&acc[(32 + rr) * 128 + cc], a2v[r]);
        atomicAdd(&acc[(48 + rr) * 128 + cc], a3[r]);
    }
}

__global__ __launch_bounds__(256)
void zero_kernel(float* __restrict__ p, int n)
{
    int i = blockIdx.x * 256 + threadIdx.x;
    if (i < n) p[i] = 0.f;
}

__global__ __launch_bounds__(256)
void clf_final_kernel(const float* __restrict__ acc, const float* __restrict__ b1,
                      const float* __restrict__ w2, const float* __restrict__ b2,
                      float* __restrict__ out)
{
    int tid = threadIdx.x;           // 256 = 64*4
    int b = tid >> 2, c = tid & 3;
    float s = b2[c];
    for (int j = 0; j < 128; ++j)
        s += fmaxf(acc[b * 128 + j] + b1[j], 0.f) * w2[j * 4 + c];
    out[b * 4 + c] = s;
}

// ---------------------------------------------------------------------------
extern "C" void kernel_launch(void* const* d_in, const int* in_sizes, int n_in,
                              void* d_out, int out_size, void* d_ws, size_t ws_size,
                              hipStream_t stream)
{
    const float* x_tf  = (const float*)d_in[0];
    const float* cw    = (const float*)d_in[1];
    const float* cb    = (const float*)d_in[2];
    const float* sa_w1 = (const float*)d_in[3];
    const float* sa_b1 = (const float*)d_in[4];
    const float* sa_w2 = (const float*)d_in[5];
    const float* sa_b2 = (const float*)d_in[6];
    const float* ta_w1 = (const float*)d_in[7];
    const float* ta_b1 = (const float*)d_in[8];
    const float* ta_w2 = (const float*)d_in[9];
    const float* ta_b2 = (const float*)d_in[10];
    const float* ln_g  = (const float*)d_in[11];
    const float* ln_b  = (const float*)d_in[12];
    const float* bn_g  = (const float*)d_in[13];
    const float* bn_b  = (const float*)d_in[14];
    const float* Wg    = (const float*)d_in[15];
    const float* bWg   = (const float*)d_in[16];
    const float* Ug    = (const float*)d_in[17];
    const float* bUg   = (const float*)d_in[18];
    const float* bg    = (const float*)d_in[19];
    const float* Wq    = (const float*)d_in[20];
    const float* bq    = (const float*)d_in[21];
    const float* Wv    = (const float*)d_in[22];
    const float* bv    = (const float*)d_in[23];
    const float* Aa    = (const float*)d_in[24];
    const float* Bw    = (const float*)d_in[25];
    const float* Bb    = (const float*)d_in[26];
    const float* Wo    = (const float*)d_in[27];
    const float* bWo   = (const float*)d_in[28];
    const float* cw1   = (const float*)d_in[29];
    const float* cb1   = (const float*)d_in[30];
    const float* cw2   = (const float*)d_in[31];
    const float* cb2   = (const float*)d_in[32];
    float* out = (float*)d_out;

    // workspace layout
    char* p = (char*)d_ws;
    auto take = [&](size_t bytes) {
        char* r = p;
        p += (bytes + 255) & ~(size_t)255;
        return r;
    };
    float*  bufY    = (float*) take((size_t)BB * CB * TT * 5 * 4);  // y / out0
    float*  bufXn   = (float*) take((size_t)TT * BB * DD * 4);      // xn / out1
    __bf16* bufXnh  = (__bf16*)take((size_t)TT * BB * DD * 2);
    float*  bufProj = (float*) take((size_t)TT * BB * 256 * 4);
    __bf16* bufH2h  = (__bf16*)take((size_t)TT * BB * SS * 2);
    __bf16* wcath   = (__bf16*)take((size_t)DD * 256 * 2);
    float*  biascat = (float*) take(256 * 4);
    __bf16* woh     = (__bf16*)take((size_t)SS * DD * 2);
    float*  avg_c   = (float*) take((size_t)BB * CB * 4);
    float*  wc      = (float*) take((size_t)BB * CB * 4);
    float*  avg_t   = (float*) take((size_t)BB * TT * 4);
    float*  wt      = (float*) take((size_t)BB * TT * 4);
    float*  accC    = (float*) take((size_t)BB * 128 * 4);

    // 1-5: conv + attentions
    {
        size_t n = (size_t)BB * CB * TT * 5;
        conv_kernel<<<(unsigned)((n + 255) / 256), 256, 0, stream>>>(x_tf, cw, cb, bufY);
    }
    avgc_kernel<<<BB * CB, 256, 0, stream>>>(bufY, avg_c);
    samlp_kernel<<<BB, 128, 0, stream>>>(avg_c, sa_w1, sa_b1, sa_w2, sa_b2, wc);
    avgt_kernel<<<BB * TT, 256, 0, stream>>>(bufY, wc, avg_t);
    tamlp_kernel<<<BB, 256, 0, stream>>>(avg_t, ta_w1, ta_b1, ta_w2, ta_b2, wt);

    // RMSU layers
    for (int i = 0; i < 2; ++i) {
        const float* WgL = Wg + (size_t)i * DD * SS;
        const float* WqL = Wq + (size_t)i * DD * SS;
        const float* WvL = Wv + (size_t)i * DD * SS;
        const float* BwL = Bw + (size_t)i * DD * SS;
        const float* WoL = Wo + (size_t)i * SS * DD;
        pack_wcat_kernel<<<(DD * 256 + 255) / 256, 256, 0, stream>>>(
            WgL, WqL, WvL, BwL,
            bWg + i * SS, bUg + i * SS, bg + i * SS, bq + i * SS, bv + i * SS,
            wcath, biascat);
        pack_bf16_kernel<<<(SS * DD + 255) / 256, 256, 0, stream>>>(WoL, woh, SS * DD);

        // LN: layer0 fuses feat = y*wc*wt; layer1 reads out0 (reused bufY)
        ln_kernel<<<TT * BB, 256, 0, stream>>>(
            bufY, wc, wt,
            ln_g + i * DD, ln_b + i * DD, bn_g + i * DD, bn_b + i * DD,
            bufXn, bufXnh, i == 0 ? 0 : 1);

        // proj = xn @ Wcat + biascat   [64000,550]x[550,256]
        gemm_bf16_kernel<<<(TT * BB / 64) * 4, 256, 0, stream>>>(
            bufXnh, wcath, biascat, nullptr, bufProj,
            TT * BB, 256, DD, 0);

        // sequential recurrence -> H2
        scan_kernel<<<1, 512, 0, stream>>>(
            bufProj, Ug + (size_t)i * SS * SS, Aa + i * SS, Bb + i * SS, bufH2h);

        // out = sigmoid(H2 @ Wo + bWo) + xn   [64000,64]x[64,550]
        float* dst = (i == 0) ? bufY : bufXn;   // layer1 writes in-place over xn
        gemm_bf16_kernel<<<(TT * BB / 64) * 9, 256, 0, stream>>>(
            bufH2h, woh, bWo + (size_t)i * DD, bufXn, dst,
            TT * BB, DD, SS, 1);
    }

    // classifier
    zero_kernel<<<(BB * 128 + 255) / 256, 256, 0, stream>>>(accC, BB * 128);
    clf_partial_kernel<<<275, 256, 0, stream>>>(bufXn, cw1, accC);
    clf_final_kernel<<<1, 256, 0, stream>>>(accC, cb1, cw2, cb2, out);
    (void)in_sizes; (void)n_in; (void)out_size; (void)ws_size;
}
